// CSCT_Engine_58428735095680
// MI455X (gfx1250) — matrix operand order repaired and checked by reading, not compiled, not run on hardware
//
#include <hip/hip_runtime.h>

#define DEV static __device__ __forceinline__

typedef __attribute__((ext_vector_type(16))) _Float16 v16h;
typedef __attribute__((ext_vector_type(8)))  _Float16 v8h;
typedef __attribute__((ext_vector_type(8)))  float    v8f;

union AFrag { v16h v; v8h h[2]; _Float16 e[16]; };

static constexpr int NTOK     = 131072;        // B*T = 16*8192
static constexpr int TLEN     = 8192;          // T
static constexpr int NTILE    = NTOK / 16;     // 8192 token tiles of 16
static constexpr int NB_STATS = 128;

static constexpr int WFRAGS      = 126;            // 36 (W1) + 72 (W2) + 18 (W3)
static constexpr int LDSW_HALVES = WFRAGS * 512;   // 64512 halves, 1KB per fragment
static constexpr int HBUF_HALVES = 2048;           // per-wave: two 16x64 f16 buffers

DEV float rcp_fast(float x) { return __builtin_amdgcn_rcpf(x); }
DEV float sigmoid_fast(float x) { return rcp_fast(1.0f + __expf(-x)); }
DEV float clampf_(float x, float lo, float hi) { return fminf(fmaxf(x, lo), hi); }

// branchless tanh: sign(x) * (1 - e) / (1 + e), e = exp(-2|x|).
// Single v_exp_f32 + v_rcp_f32, no EXEC divergence (libm tanhf branches per lane).
DEV float tanh_fast(float x) {
  float e = __expf(-2.0f * fabsf(x));
  float t = (1.0f - e) * rcp_fast(1.0f + e);
  return copysignf(t, x);
}

// LDS ops are in-order within a wave (DScnt); this guarantees completion and
// stops the compiler from reordering ds ops across the staging point.
DEV void lds_fence() { __asm__ volatile("s_wait_dscnt 0" ::: "memory"); }

// raw (unscaled) [y, dy, d2y] for one D=2 signal at token n (tt = n % T)
DEV void raw6(const float2* __restrict__ p, int n, int tt, float* o) {
  float2 a = p[n];
  float2 b = (tt >= 1) ? p[n - 1] : a;   // dy  = 0 at tt==0
  float2 c = (tt >= 2) ? p[n - 2] : b;   // d2y = 0 at tt==0, = dy at tt==1
  o[0] = a.x;               o[1] = a.y;
  o[2] = a.x - b.x;         o[3] = a.y - b.y;
  o[4] = a.x - 2.f*b.x + c.x; o[5] = a.y - 2.f*b.y + c.y;
}

// ---------------- Kernel 1: per-block partial sums of |f| and f^2 ----------------
__global__ void __launch_bounds__(256) k_stats(const float* __restrict__ xt,
                                               const float* __restrict__ ya,
                                               float* __restrict__ partials) {
  float sA[12], sS[12];
#pragma unroll
  for (int i = 0; i < 12; ++i) { sA[i] = 0.f; sS[i] = 0.f; }
  const float2* px = (const float2*)xt;
  const float2* py = (const float2*)ya;
  for (int n = blockIdx.x * blockDim.x + threadIdx.x; n < NTOK; n += gridDim.x * blockDim.x) {
    int tt = n & (TLEN - 1);
    float r[12];
    raw6(px, n, tt, r);
    raw6(py, n, tt, r + 6);
#pragma unroll
    for (int i = 0; i < 12; ++i) { float v = r[i]; sA[i] += fabsf(v); sS[i] += v * v; }
  }
#pragma unroll
  for (int i = 0; i < 12; ++i) {
    for (int s = 16; s > 0; s >>= 1) {
      sA[i] += __shfl_xor(sA[i], s, 32);
      sS[i] += __shfl_xor(sS[i], s, 32);
    }
  }
  __shared__ float wsum[8][24];
  int lane = threadIdx.x & 31, wave = threadIdx.x >> 5;
  if (lane == 0) {
#pragma unroll
    for (int i = 0; i < 12; ++i) { wsum[wave][i] = sA[i]; wsum[wave][12 + i] = sS[i]; }
  }
  __syncthreads();
  int c = threadIdx.x;
  if (c < 24) {
    float s = 0.f;
    int nw = blockDim.x >> 5;
    for (int w = 0; w < nw; ++w) s += wsum[w][c];
    partials[blockIdx.x * 24 + c] = s;   // deterministic (no atomics)
  }
}

// ---------------- Kernel 2: final deterministic reduction ----------------
__global__ void k_reduce(const float* __restrict__ partials, float* __restrict__ stats) {
  int c = threadIdx.x;
  if (c < 24) {
    float s = 0.f;
    for (int i = 0; i < NB_STATS; ++i) s += partials[i * 24 + c];
    stats[c] = s;
  }
}

// ---------------- Kernel 3: gates + argmax -> kstar ----------------
__global__ void __launch_bounds__(256) k_gate(
    const float* __restrict__ xt, const float* __restrict__ ya,
    const float* __restrict__ naW1, const float* __restrict__ nab1,
    const float* __restrict__ naW2, const float* __restrict__ nab2,
    const float* __restrict__ thW,  const float* __restrict__ thb,
    const float* __restrict__ nmW1, const float* __restrict__ nmb1,
    const float* __restrict__ nmW2, const float* __restrict__ nmb2,
    const float* __restrict__ p_nathr, const float* __restrict__ p_nmthr,
    const float* __restrict__ p_sens,  const float* __restrict__ p_tfreq,
    const float* __restrict__ stats, unsigned char* __restrict__ kstar)
{
  __shared__ float sW[2752];  // naW1(768) | naW2(576) | nmW1(832) | nmW2(576)
  for (int i = threadIdx.x; i < 768; i += blockDim.x) sW[i]        = naW1[i];
  for (int i = threadIdx.x; i < 576; i += blockDim.x) sW[768 + i]  = naW2[i];
  for (int i = threadIdx.x; i < 832; i += blockDim.x) sW[1344 + i] = nmW1[i];
  for (int i = threadIdx.x; i < 576; i += blockDim.x) sW[2176 + i] = nmW2[i];
  __syncthreads();

  int n = blockIdx.x * blockDim.x + threadIdx.x;   // grid sized exactly
  int tt = n & (TLEN - 1);

  float fs[12];
  {
    float r[12];
    raw6((const float2*)xt, n, tt, r);
    raw6((const float2*)ya, n, tt, r + 6);
    const float invN = 1.0f / (float)NTOK;
#pragma unroll
    for (int c = 0; c < 12; ++c) {
      float m1 = stats[c] * invN, m2 = stats[12 + c] * invN;
      float var = m2 - m1 * m1; var = var > 0.f ? var : 0.f;
      float sd = sqrtf(var); sd = sd < 1e-5f ? 1e-5f : sd;
      fs[c] = clampf_(r[c] / sd, -3.f, 3.f);
    }
  }
  float str2 = 0.f;
#pragma unroll
  for (int c = 0; c < 12; ++c) str2 += fs[c] * fs[c];
  float strength = sqrtf(str2);
  float sc = clampf_(p_sens[0], 0.f, 1.f);
  float tf = clampf_(p_tfreq[0], 0.1f, 16.f);

  // Na+ gate MLP
  float t1[64];
#pragma unroll
  for (int h = 0; h < 64; ++h) {
    float a = nab1[h];
#pragma unroll
    for (int c = 0; c < 12; ++c) a = fmaf(fs[c], sW[c * 64 + h], a);
    t1[h] = tanh_fast(a);
  }
  float thr_na = p_nathr[0] - sc * strength;
  float naact[9];
#pragma unroll
  for (int j = 0; j < 9; ++j) {
    float a = nab2[j];
#pragma unroll
    for (int h = 0; h < 64; ++h) a = fmaf(t1[h], sW[768 + h * 9 + j], a);
    a = clampf_(a, -10.f, 10.f);
    naact[j] = (a > thr_na) ? sigmoid_fast(a) : 0.f;   // hard mask * sigmoid
  }

  // theta phase / window
  float tm = thb[0];
#pragma unroll
  for (int c = 0; c < 12; ++c) tm = fmaf(fs[c], thW[c], tm);
  tm = sigmoid_fast(tm);
  float tl = (float)tt * (1.0f / 8191.0f);   // linspace(0,1,T)
  float phase = clampf_(6.28318530717958647692f * tf * tl * tm, -100.f, 100.f);
  float window = sigmoid_fast(__sinf(phase) * 4.0f);  // |phase|<=100 << v_sin range

  // NMDA gate MLP (input = feat ++ phase)
#pragma unroll
  for (int h = 0; h < 64; ++h) {
    float a = nmb1[h];
#pragma unroll
    for (int c = 0; c < 12; ++c) a = fmaf(fs[c], sW[1344 + c * 64 + h], a);
    a = fmaf(phase, sW[1344 + 768 + h], a);
    t1[h] = tanh_fast(a);
  }
  float thr_n = p_nmthr[0] - sc * strength;

  int best = 0; float bg = 0.f;
#pragma unroll
  for (int j = 0; j < 9; ++j) {
    float a = nmb2[j];
#pragma unroll
    for (int h = 0; h < 64; ++h) a = fmaf(t1[h], sW[2176 + h * 9 + j], a);
    a = clampf_(a, -10.f, 10.f);
    float nm = (a > thr_n) ? window * sigmoid_fast(a) : 0.f;
    float g = naact[j] * nm;
    if (j == 0) { bg = g; }
    else if (g > bg) { bg = g; best = j; }   // first-max = jnp.argmax
  }
  kstar[n] = (unsigned char)best;
}

// ---------------- Kernel 4: 9 clock experts via WMMA + top-1 select ----------------
__global__ void __launch_bounds__(128) k_experts(
    const float* __restrict__ z,
    const float* __restrict__ eW1, const float* __restrict__ eb1,
    const float* __restrict__ eW2, const float* __restrict__ eb2,
    const float* __restrict__ eW3, const float* __restrict__ eb3,
    const float* __restrict__ cbk, const float* __restrict__ bias,
    const unsigned char* __restrict__ kstar,
    float* __restrict__ out)
{
  extern __shared__ _Float16 smem[];

  // ---- build pre-swizzled B fragments in LDS (once per workgroup) ----
  // fragment layout: smem[f*512 + lane*16 + e], B element (K = 8*(lane>>4)+(e&7)+16*(e>>3) [+32*s], N = lane&15 [+16*c])
  for (int p = threadIdx.x; p < WFRAGS * 32; p += blockDim.x) {
    int f = p >> 5, L = p & 31;
    int lhi = L >> 4, llo = L & 15;
    _Float16* dst = &smem[f * 512 + L * 16];
    if (f < 36) {                       // eW1[k]: 16x64, K padded 16->32
      int k = f >> 2, c = f & 3;
#pragma unroll
      for (int e = 0; e < 16; ++e) {
        int Kd = lhi * 8 + (e & 7) + ((e >> 3) << 4);
        float v = (Kd < 16) ? eW1[(k * 16 + Kd) * 64 + c * 16 + llo] : 0.f;
        dst[e] = (_Float16)v;
      }
    } else if (f < 108) {               // eW2[k]: 64x64, 2 K-slabs x 4 N-tiles
      int g = f - 36; int k = g >> 3, c = (g >> 1) & 3, s = g & 1;
#pragma unroll
      for (int e = 0; e < 16; ++e) {
        int Kd = s * 32 + lhi * 8 + (e & 7) + ((e >> 3) << 4);
        dst[e] = (_Float16)eW2[(k * 64 + Kd) * 64 + c * 16 + llo];
      }
    } else {                            // eW3[k]: 64x2, N padded 2->16
      int g = f - 108; int k = g >> 1, s = g & 1;
#pragma unroll
      for (int e = 0; e < 16; ++e) {
        int Kd = s * 32 + lhi * 8 + (e & 7) + ((e >> 3) << 4);
        float v = (llo < 2) ? eW3[(k * 64 + Kd) * 2 + llo] : 0.f;
        dst[e] = (_Float16)v;
      }
    }
  }
  __syncthreads();

  const int lane = threadIdx.x & 31;
  const int wave = threadIdx.x >> 5;
  const int rowM = lane & 15;           // A-row / D-column owned by this lane
  const int hb8  = (lane >> 4) * 8;     // K/M half selector
  _Float16* hA = &smem[LDSW_HALVES + wave * HBUF_HALVES];  // 16x64 f16, row-major
  _Float16* hB = hA + 1024;
  const bool sel = (rowM < 2);          // lanes 0,1,16,17 own output columns d=0,1

  const int wavesTotal = gridDim.x * (blockDim.x >> 5);
  for (int tile = blockIdx.x * (blockDim.x >> 5) + wave; tile < NTILE; tile += wavesTotal) {
    // prefetch next tile's inputs (global_prefetch_b8); address known 9 GEMMs ahead
    {
      int nt = tile + wavesTotal;
      if (nt < NTILE) {
        __builtin_prefetch(z + (size_t)(nt * 16 + rowM) * 16 + hb8, 0, 1);
        __builtin_prefetch(kstar + nt * 16, 0, 1);
      }
    }
    // A operand for layer 1 straight from global z (16 real K, zero-padded to 32)
    AFrag az;
    const float4* zp = (const float4*)(z + (size_t)(tile * 16 + rowM) * 16 + hb8);
    float4 z0 = zp[0], z1 = zp[1];
    az.e[0] = (_Float16)z0.x; az.e[1] = (_Float16)z0.y; az.e[2] = (_Float16)z0.z; az.e[3] = (_Float16)z0.w;
    az.e[4] = (_Float16)z1.x; az.e[5] = (_Float16)z1.y; az.e[6] = (_Float16)z1.z; az.e[7] = (_Float16)z1.w;
#pragma unroll
    for (int e = 8; e < 16; ++e) az.e[e] = (_Float16)0.f;

    int ks[8]; float outv[8];
    if (sel) {
#pragma unroll
      for (int r = 0; r < 8; ++r) { ks[r] = kstar[tile * 16 + hb8 + r]; outv[r] = 0.f; }
    }

    for (int k = 0; k < 9; ++k) {
      // ---- layer 1: z(16x16) @ eW1 -> 16x64 ----
      v8f acc1[4];
#pragma unroll
      for (int c4 = 0; c4 < 4; ++c4) {
        v16h b = *(const v16h*)&smem[(k * 4 + c4) * 512 + lane * 16];
        v8f zc = {};
        acc1[c4] = __builtin_amdgcn_wmma_f32_16x16x32_f16(false, az.v, false, b, (short)0, zc, false, false);
      }
#pragma unroll
      for (int c4 = 0; c4 < 4; ++c4) {
        float b1v = eb1[k * 64 + c4 * 16 + rowM];
#pragma unroll
        for (int r = 0; r < 8; ++r) {
          float hv = tanh_fast(acc1[c4][r] + b1v);
          hA[(hb8 + r) * 64 + c4 * 16 + rowM] = (_Float16)hv;   // h1[M][N]
        }
      }
      lds_fence();
      AFrag a10, a11;   // 16x64 A operand as two K-slabs
      a10.h[0] = *(const v8h*)&hA[rowM * 64 +  0 + hb8];
      a10.h[1] = *(const v8h*)&hA[rowM * 64 + 16 + hb8];
      a11.h[0] = *(const v8h*)&hA[rowM * 64 + 32 + hb8];
      a11.h[1] = *(const v8h*)&hA[rowM * 64 + 48 + hb8];

      // ---- layer 2: h1(16x64) @ eW2 -> 16x64 ----
#pragma unroll
      for (int c4 = 0; c4 < 4; ++c4) {
        v16h b0 = *(const v16h*)&smem[(36 + k * 8 + c4 * 2 + 0) * 512 + lane * 16];
        v16h b1 = *(const v16h*)&smem[(36 + k * 8 + c4 * 2 + 1) * 512 + lane * 16];
        v8f zc = {};
        v8f t = __builtin_amdgcn_wmma_f32_16x16x32_f16(false, a10.v, false, b0, (short)0, zc, false, false);
        t     = __builtin_amdgcn_wmma_f32_16x16x32_f16(false, a11.v, false, b1, (short)0, t,  false, false);
        float b2v = eb2[k * 64 + c4 * 16 + rowM];
#pragma unroll
        for (int r = 0; r < 8; ++r) {
          float hv = tanh_fast(t[r] + b2v);
          hB[(hb8 + r) * 64 + c4 * 16 + rowM] = (_Float16)hv;   // h2[M][N]
        }
      }
      lds_fence();
      AFrag a20, a21;
      a20.h[0] = *(const v8h*)&hB[rowM * 64 +  0 + hb8];
      a20.h[1] = *(const v8h*)&hB[rowM * 64 + 16 + hb8];
      a21.h[0] = *(const v8h*)&hB[rowM * 64 + 32 + hb8];
      a21.h[1] = *(const v8h*)&hB[rowM * 64 + 48 + hb8];

      // ---- layer 3: h2(16x64) @ eW3 -> 16x2 (N padded to 16) ----
      v16h w30 = *(const v16h*)&smem[(108 + k * 2 + 0) * 512 + lane * 16];
      v16h w31 = *(const v16h*)&smem[(108 + k * 2 + 1) * 512 + lane * 16];
      v8f zc3 = {};
      v8f u = __builtin_amdgcn_wmma_f32_16x16x32_f16(false, a20.v, false, w30, (short)0, zc3, false, false);
      u     = __builtin_amdgcn_wmma_f32_16x16x32_f16(false, a21.v, false, w31, (short)0, u,   false, false);

      if (sel) {        // lane owns output column d = rowM; VGPR r -> token row hb8+r
        float ebc = eb3[k * 2 + rowM] + cbk[k * 2 + rowM];
#pragma unroll
        for (int r = 0; r < 8; ++r) outv[r] = (ks[r] == k) ? (u[r] + ebc) : outv[r];
      }
      lds_fence();
    }

    if (sel) {
      float bd = bias[rowM];
#pragma unroll
      for (int r = 0; r < 8; ++r)
        out[(size_t)(tile * 16 + hb8 + r) * 2 + rowM] = outv[r] + bd;
    }
  }
}

extern "C" void kernel_launch(void* const* d_in, const int* in_sizes, int n_in,
                              void* d_out, int out_size, void* d_ws, size_t ws_size,
                              hipStream_t stream) {
  (void)in_sizes; (void)n_in; (void)out_size; (void)ws_size;
  const float* xt    = (const float*)d_in[0];
  const float* ya    = (const float*)d_in[1];
  const float* z     = (const float*)d_in[2];
  const float* naW1  = (const float*)d_in[3];
  const float* nab1  = (const float*)d_in[4];
  const float* naW2  = (const float*)d_in[5];
  const float* nab2  = (const float*)d_in[6];
  const float* thW   = (const float*)d_in[7];
  const float* thb   = (const float*)d_in[8];
  const float* nmW1  = (const float*)d_in[9];
  const float* nmb1  = (const float*)d_in[10];
  const float* nmW2  = (const float*)d_in[11];
  const float* nmb2  = (const float*)d_in[12];
  const float* nathr = (const float*)d_in[13];
  const float* nmthr = (const float*)d_in[14];
  const float* sens  = (const float*)d_in[15];
  // d_in[16] = gate_temp: cancels exactly in the forward pass (soft - stopgrad(soft) == 0)
  const float* tfreq = (const float*)d_in[17];
  const float* eW1   = (const float*)d_in[18];
  const float* eb1   = (const float*)d_in[19];
  const float* eW2   = (const float*)d_in[20];
  const float* eb2   = (const float*)d_in[21];
  const float* eW3   = (const float*)d_in[22];
  const float* eb3   = (const float*)d_in[23];
  const float* cbk   = (const float*)d_in[24];
  const float* bias  = (const float*)d_in[25];

  char* ws = (char*)d_ws;
  float*         stats    = (float*)ws;                       // 24 floats
  unsigned char* kstar    = (unsigned char*)(ws + 128);       // 131072 bytes
  float*         partials = (float*)(ws + 128 + 131072);      // 128*24 floats

  k_stats<<<NB_STATS, 256, 0, stream>>>(xt, ya, partials);
  k_reduce<<<1, 32, 0, stream>>>(partials, stats);
  k_gate<<<NTOK / 256, 256, 0, stream>>>(xt, ya, naW1, nab1, naW2, nab2, thW, thb,
                                         nmW1, nmb1, nmW2, nmb2, nathr, nmthr, sens, tfreq,
                                         stats, kstar);
  size_t ldsBytes = (size_t)(LDSW_HALVES + 4 * HBUF_HALVES) * 2;  // ~142 KB of the 320 KB WGP LDS
  k_experts<<<256, 128, ldsBytes, stream>>>(z, eW1, eb1, eW2, eb2, eW3, eb3, cbk, bias,
                                            kstar, (float*)d_out);
}
